// ChannelGroupAttention_2087354106724
// MI455X (gfx1250) — matrix operand order repaired and checked by reading, not compiled, not run on hardware
//
#include <hip/hip_runtime.h>
#include <hip/hip_bf16.h>

typedef __attribute__((ext_vector_type(16))) __bf16 v16bf;
typedef __attribute__((ext_vector_type(8)))  float  v8f;

#define Bb 8
#define Pp 512
#define Cc 1024
#define Hh 8
#define Ww 64

union Frag {
    v16bf v;
    unsigned short s[16];
    unsigned u[8];
};

// RNE f32->bf16, scalar (epilogues only)
static __device__ __forceinline__ unsigned short f2bf(float f) {
    unsigned u = __float_as_uint(f);
    u += 0x7FFFu + ((u >> 16) & 1u);
    return (unsigned short)(u >> 16);
}

// RNE f32x2 -> packed bf16x2 via v_perm_b32 (low = f0, high = f1)
static __device__ __forceinline__ unsigned pack_bf16_rne(float f0, float f1) {
    unsigned u0 = __float_as_uint(f0);
    unsigned u1 = __float_as_uint(f1);
    u0 += 0x7FFFu + ((u0 >> 16) & 1u);
    u1 += 0x7FFFu + ((u1 >> 16) & 1u);
    // bytes {u1[3],u1[2],u0[3],u0[2]} -> takes [31:16] of each rounded value
    return __builtin_amdgcn_perm(u1, u0, 0x07060302u);
}

// A-fragment K index (16-bit A 16x32, ISA 7.12.2): interleaved halves
static __device__ __forceinline__ int kA(int half, int j) {
    return (j < 4) ? (half * 8 + 2 * j) : (16 + half * 8 + 2 * (j - 4));
}
// B-fragment K index (16-bit B 32x16): interleaved halves (same as A)
static __device__ __forceinline__ int kB(int half, int j) {
    return (j < 4) ? (half * 8 + 2 * j) : (16 + half * 8 + 2 * (j - 4));
}

// ---------------------------------------------------------------------------
// Kernel 0: bulk f32 -> bf16 conversion (float4 in, uint2 out), RNE.
// ---------------------------------------------------------------------------
__global__ void __launch_bounds__(256)
k_cvt_bf16(const float* __restrict__ in, unsigned short* __restrict__ outp, int n4)
{
    const int i = blockIdx.x * 256 + threadIdx.x;
    if (i >= n4) return;
    const float4 v = ((const float4*)in)[i];
    uint2 o;
    o.x = pack_bf16_rne(v.x, v.y);
    o.y = pack_bf16_rne(v.z, v.w);
    ((uint2*)outp)[i] = o;
}

// ---------------------------------------------------------------------------
// Kernel 1: qkv = x @ W^T (bf16 operands), fused BatchNorm.
// One wave computes a 16x64 tile (4 accumulators, A reused 4x).
// Writes q[b][h][d][w] (bf16) and transposed qT[b][h][w][d] (bf16).
// ---------------------------------------------------------------------------
__global__ void __launch_bounds__(128)
k_qkv_bn(const unsigned short* __restrict__ xbf,
         const unsigned short* __restrict__ Wbf,
         const float* __restrict__ gamma, const float* __restrict__ beta,
         const float* __restrict__ mean, const float* __restrict__ var,
         unsigned short* __restrict__ q, unsigned short* __restrict__ qT)
{
    const int wave = blockIdx.x * 4 + (threadIdx.x >> 5);
    const int lane = threadIdx.x & 31;
    const int half = lane >> 4;
    const int lm   = lane & 15;

    const int tm = wave / (Cc / 64);      // 256 row tiles over B*P
    const int tn = wave % (Cc / 64);      // 16 tiles of 64 channels
    const int rowA = tm * 16 + lm;

    v8f acc[4] = {{}, {}, {}, {}};
    for (int kk = 0; kk < Cc; kk += 32) {
        Frag a;
#pragma unroll
        for (int j = 0; j < 8; ++j)
            a.u[j] = *(const unsigned*)(xbf + (size_t)rowA * Cc + kk + kA(half, j));
#pragma unroll
        for (int s = 0; s < 4; ++s) {
            const int n = tn * 64 + s * 16 + lm;
            Frag b;
#pragma unroll
            for (int j = 0; j < 8; ++j)
                b.u[j] = *(const unsigned*)(Wbf + (size_t)n * Cc + kk + kB(half, j));
            acc[s] = __builtin_amdgcn_wmma_f32_16x16x32_bf16(
                false, a.v, false, b.v, (short)0, acc[s], false, false);
        }
    }

    // C/D layout: VGPR r -> row m = tm*16 + half*8 + r ; tile fits one (b,h).
    const int row0 = tm * 16 + half * 8;
    const int bat  = row0 >> 9;
    const int p0   = row0 & 511;
    const int hh   = p0 >> 6;
    const int w0   = p0 & 63;             // multiple of 8
    const size_t bhOff = (size_t)bat * Hh + hh;
    const size_t qtb   = bhOff * Ww * Cc;

#pragma unroll
    for (int s = 0; s < 4; ++s) {
        const int n  = tn * 64 + s * 16 + lm;
        const float iv = rsqrtf(var[n] + 1e-5f);
        const float sc = gamma[n] * iv;
        const float mu = mean[n];
        const float be = beta[n];
        unsigned short bv[8];
#pragma unroll
        for (int r = 0; r < 8; ++r)
            bv[r] = f2bf((acc[s][r] - mu) * sc + be);
        unsigned short* qp = q + (bhOff * Cc + n) * Ww + w0;
#pragma unroll
        for (int r = 0; r < 8; r += 2)
            *(unsigned*)(qp + r) = (unsigned)bv[r] | ((unsigned)bv[r + 1] << 16);
#pragma unroll
        for (int r = 0; r < 8; ++r)
            qT[qtb + (size_t)(w0 + r) * Cc + n] = bv[r];
    }
}

// ---------------------------------------------------------------------------
// Kernel 2: scores[b,h,c,d] = (q . q^T) / gra_sharp -> attn region (raw).
// One wave computes a 16x32 tile (2 accumulators, A reused 2x). K=64.
// ---------------------------------------------------------------------------
__global__ void __launch_bounds__(128)
k_scores(const unsigned short* __restrict__ q, const float* __restrict__ gra,
         float* __restrict__ attn)
{
    const int wave = blockIdx.x * 4 + (threadIdx.x >> 5);
    const int lane = threadIdx.x & 31;
    const int half = lane >> 4;
    const int lm   = lane & 15;

    const int tilesPerBH = (Cc / 16) * (Cc / 32);   // 64 * 32 = 2048
    const int bh = wave / tilesPerBH;
    const int t  = wave % tilesPerBH;
    const int tc = t / (Cc / 32);
    const int td = t % (Cc / 32);

    const size_t qb = (size_t)bh * Cc * Ww;
    const int rc  = tc * 16 + lm;        // A row (c)
    const int rd0 = td * 32 + lm;        // B cols (d) -> contiguous q rows
    const int rd1 = rd0 + 16;

    v8f acc0 = {}, acc1 = {};
#pragma unroll
    for (int kk = 0; kk < Ww; kk += 32) {
        Frag a, b0, b1;
#pragma unroll
        for (int j = 0; j < 8; ++j) {
            const int ka = kk + kA(half, j);
            const int kb = kk + kB(half, j);
            a.u[j]  = *(const unsigned*)(q + qb + (size_t)rc  * Ww + ka);
            b0.u[j] = *(const unsigned*)(q + qb + (size_t)rd0 * Ww + kb);
            b1.u[j] = *(const unsigned*)(q + qb + (size_t)rd1 * Ww + kb);
        }
        acc0 = __builtin_amdgcn_wmma_f32_16x16x32_bf16(
            false, a.v, false, b0.v, (short)0, acc0, false, false);
        acc1 = __builtin_amdgcn_wmma_f32_16x16x32_bf16(
            false, a.v, false, b1.v, (short)0, acc1, false, false);
    }

    const float is = 1.0f / gra[0];
    float* op = attn + (size_t)bh * Cc * Cc
              + (size_t)(tc * 16 + half * 8) * Cc;
#pragma unroll
    for (int r = 0; r < 8; ++r) {
        op[(size_t)r * Cc + rd0] = acc0[r] * is;
        op[(size_t)r * Cc + rd1] = acc1[r] * is;
    }
}

// ---------------------------------------------------------------------------
// Kernel 3: softmax over the HEAD axis (axis=1), in place, float4 vectorized.
// ---------------------------------------------------------------------------
__global__ void __launch_bounds__(256)
k_softmax_h(float* __restrict__ attn)
{
    const size_t CC4   = (size_t)Cc * Cc / 4;
    const size_t total = (size_t)Bb * CC4;
    const size_t i = (size_t)blockIdx.x * blockDim.x + threadIdx.x;
    if (i >= total) return;

    const size_t b   = i / CC4;
    const size_t cd4 = i % CC4;
    float4* base = (float4*)attn + b * Hh * CC4 + cd4;

    float4 s[Hh];
#pragma unroll
    for (int h = 0; h < Hh; ++h) s[h] = base[(size_t)h * CC4];

    float4 m = s[0];
#pragma unroll
    for (int h = 1; h < Hh; ++h) {
        m.x = fmaxf(m.x, s[h].x); m.y = fmaxf(m.y, s[h].y);
        m.z = fmaxf(m.z, s[h].z); m.w = fmaxf(m.w, s[h].w);
    }
    float4 sum = make_float4(0.f, 0.f, 0.f, 0.f);
#pragma unroll
    for (int h = 0; h < Hh; ++h) {
        s[h].x = __expf(s[h].x - m.x); sum.x += s[h].x;
        s[h].y = __expf(s[h].y - m.y); sum.y += s[h].y;
        s[h].z = __expf(s[h].z - m.z); sum.z += s[h].z;
        s[h].w = __expf(s[h].w - m.w); sum.w += s[h].w;
    }
    const float4 r = make_float4(1.f / sum.x, 1.f / sum.y,
                                 1.f / sum.z, 1.f / sum.w);
#pragma unroll
    for (int h = 0; h < Hh; ++h) {
        float4 o;
        o.x = s[h].x * r.x; o.y = s[h].y * r.y;
        o.z = s[h].z * r.z; o.w = s[h].w * r.w;
        base[(size_t)h * CC4] = o;
    }
}

// ---------------------------------------------------------------------------
// Kernel 4: h[b,h,c,w] = attn @ q. One wave computes 16 x 64 (full N):
// A (attn row, f32) converted once per K-step and reused across 4 WMMAs;
// B fragments come from contiguous bf16 rows of qT. Output written merged
// as out[b, p = h*64 + w, c] with two b128 stores per lane per sub-tile.
// ---------------------------------------------------------------------------
__global__ void __launch_bounds__(128)
k_out(const float* __restrict__ attn, const unsigned short* __restrict__ qT,
      float* __restrict__ out)
{
    const int wave = blockIdx.x * 4 + (threadIdx.x >> 5);
    const int lane = threadIdx.x & 31;
    const int half = lane >> 4;
    const int lm   = lane & 15;

    const int bh = wave / (Cc / 16);     // 64
    const int tc = wave % (Cc / 16);     // 64 row tiles
    const size_t ab  = (size_t)bh * Cc * Cc;
    const size_t qtb = (size_t)bh * Ww * Cc;
    const int rc = tc * 16 + lm;

    v8f acc[4] = {{}, {}, {}, {}};
    for (int kk = 0; kk < Cc; kk += 32) {
        Frag a;
#pragma unroll
        for (int j = 0; j < 8; ++j) {
            const int ka = kk + kA(half, j);
            const float2 av = *(const float2*)(attn + ab + (size_t)rc * Cc + ka);
            a.u[j] = pack_bf16_rne(av.x, av.y);
        }
#pragma unroll
        for (int s = 0; s < 4; ++s) {
            const int n = s * 16 + lm;   // output w
            Frag b;
#pragma unroll
            for (int j = 0; j < 8; ++j)
                b.u[j] = *(const unsigned*)(qT + qtb + (size_t)n * Cc + kk + kB(half, j));
            acc[s] = __builtin_amdgcn_wmma_f32_16x16x32_bf16(
                false, a.v, false, b.v, (short)0, acc[s], false, false);
        }
    }

    const int bat = bh >> 3;
    const int hh  = bh & 7;
    const int c0  = tc * 16 + half * 8;
#pragma unroll
    for (int s = 0; s < 4; ++s) {
        const int w = s * 16 + lm;
        float* op = out + ((size_t)bat * Pp + (size_t)hh * Ww + w) * Cc + c0;
        float4 o0 = make_float4(acc[s][0], acc[s][1], acc[s][2], acc[s][3]);
        float4 o1 = make_float4(acc[s][4], acc[s][5], acc[s][6], acc[s][7]);
        *(float4*)(op)     = o0;
        *(float4*)(op + 4) = o1;
    }
}

// ---------------------------------------------------------------------------
extern "C" void kernel_launch(void* const* d_in, const int* in_sizes, int n_in,
                              void* d_out, int out_size, void* d_ws, size_t ws_size,
                              hipStream_t stream)
{
    const float* x     = (const float*)d_in[0];
    const float* Wq    = (const float*)d_in[1];
    const float* gamma = (const float*)d_in[2];
    const float* beta  = (const float*)d_in[3];
    const float* mean  = (const float*)d_in[4];
    const float* var   = (const float*)d_in[5];
    const float* gra   = (const float*)d_in[6];

    float* out_h = (float*)d_out;                       // (B, P, C)
    float* attn  = out_h + (size_t)Bb * Pp * Cc;        // (B, H, C, C)

    const size_t nX = (size_t)Bb * Pp * Cc;             // 4,194,304
    const size_t nW = (size_t)Cc * Cc;                  // 1,048,576
    unsigned short* xbf = (unsigned short*)d_ws;        // bf16 x
    unsigned short* Wbf = xbf + nX;                     // bf16 W
    unsigned short* q   = Wbf + nW;                     // bf16 (B,H,C,Ww)
    unsigned short* qT  = q + nX;                       // bf16 (B,H,Ww,C)

    // 0) one-shot f32 -> bf16 conversion of x and W
    k_cvt_bf16<<<(int)(nX / 4 / 256), 256, 0, stream>>>(x, xbf, (int)(nX / 4));
    k_cvt_bf16<<<(int)(nW / 4 / 256), 256, 0, stream>>>(Wq, Wbf, (int)(nW / 4));

    // 1) qkv GEMM + BN -> q, qT : 256 * 16 = 4096 waves (16x64 tiles)
    k_qkv_bn<<<1024, 128, 0, stream>>>(xbf, Wbf, gamma, beta, mean, var, q, qT);

    // 2) scores (pre-softmax, scaled) : 64 * 2048 = 131072 waves (16x32 tiles)
    k_scores<<<32768, 128, 0, stream>>>(q, gra, attn);

    // 3) softmax over head axis, in place : B*C*C/4 float4 lanes
    k_softmax_h<<<(Bb * Cc * (Cc / 4)) / 256, 256, 0, stream>>>(attn);

    // 4) attn @ q -> merged-head output : 64 * 64 = 4096 waves (16x64 tiles)
    k_out<<<1024, 128, 0, stream>>>(attn, qT, out_h);
}
